// Evoformer_42786464202815
// MI455X (gfx1250) — compile-verified, matrix operand order checked by reading, not run on hardware
//
#include <hip/hip_runtime.h>

// ============================================================================
// Evoformer block for MI455X (gfx1250).
// - All contractions on v_wmma_f32_16x16x32_bf16 (bf16 operands, f32 acc).
// - Working set (~40MB) lives in the 192MB L2; HBM is not the limiter.
// - Fused triangle attention keeps the full 256x256 score tile in the 320KB
//   WGP LDS; Q/K/V staged via async LDS-DMA (ASYNCcnt) when available.
// - Wave32 tiling: 16x32 C per wave (A-tile reuse), 8 waves / 256-thread block.
// ============================================================================

typedef __bf16 bf16;
typedef __attribute__((ext_vector_type(16))) __bf16 v16bf;
typedef __attribute__((ext_vector_type(8)))  float  v8f;

#if __has_builtin(__builtin_amdgcn_global_load_async_to_lds_b128)
#define EVO_ASYNC 1
// Parameter types per hipcc diagnostic: 128-bit int vector pointers,
// src in global (AS1), dst in LDS (AS3).
typedef int evo_v4i __attribute__((vector_size(4 * sizeof(int))));
typedef __attribute__((address_space(1))) evo_v4i* evo_gp;
typedef __attribute__((address_space(3))) evo_v4i* evo_lp;
#else
#define EVO_ASYNC 0
#endif

// ---------------------------------------------------------------- helpers ---
__device__ __forceinline__ v8f vz8() {
  v8f z;
#pragma unroll
  for (int i = 0; i < 8; ++i) z[i] = 0.0f;
  return z;
}

__device__ __forceinline__ v8f wmma32(v16bf a, v16bf b, v8f c) {
  // D = A(16x32 bf16) * B(32x16 bf16) + C(f32)
  return __builtin_amdgcn_wmma_f32_16x16x32_bf16(false, a, false, b, (short)0,
                                                 c, false, false);
}

__device__ __forceinline__ float sigf(float x) { return 1.0f / (1.0f + __expf(-x)); }

// 16-bit A-matrix layout (ISA 7.12.2): lane L<16 holds row M=L, K = {0..7,16..23};
// lanes 16..31 hold K = {8..15,24..31}.  => two contiguous 16B chunks per lane.
__device__ __forceinline__ v16bf load16_gap(const bf16* p) {
  v16bf a;
  float4* d = reinterpret_cast<float4*>(&a);
  d[0] = *reinterpret_cast<const float4*>(p);
  d[1] = *reinterpret_cast<const float4*>(p + 16);
  return a;
}

// A[m][k], row-major, ld elements, k contiguous.
__device__ __forceinline__ v16bf load_tileA(const bf16* A, int ld, int m0, int k0) {
  int lane = threadIdx.x & 31;
  const bf16* p = A + (size_t)(m0 + (lane & 15)) * ld + k0 + ((lane >> 4) << 3);
  return load16_gap(p);
}

// Bt[n][k], row-major (B transposed so K contiguous) -- no bounds check.
__device__ __forceinline__ v16bf load_tileBt_nb(const bf16* Bt, int ld, int n0,
                                                int k0) {
  int lane = threadIdx.x & 31;
  const bf16* p = Bt + (size_t)(n0 + (lane & 15)) * ld + k0 + ((lane >> 4) << 3);
  return load16_gap(p);
}

// Bounds-checked Bt loader (only for the N=4/8 bias projections).
__device__ __forceinline__ v16bf load_tileBt(const bf16* Bt, int ld, int n0,
                                             int k0, int Nmax) {
  int lane = threadIdx.x & 31;
  int n = n0 + (lane & 15);
  if (n >= Nmax) {
    v16bf z;
#pragma unroll
    for (int i = 0; i < 16; ++i) z[i] = (bf16)0.0f;
    return z;
  }
  const bf16* p = Bt + (size_t)n * ld + k0 + ((lane >> 4) << 3);
  return load16_gap(p);
}

// Gather loader: element(n,k) = base[n*nstride + k*kstride].
__device__ __forceinline__ v16bf load_tile_gather(const bf16* base, long nstride,
                                                  long kstride, int n0, int k0) {
  int lane = threadIdx.x & 31;
  int kh = lane >> 4;
  const bf16* p = base + (long)(n0 + (lane & 15)) * nstride;
  v16bf a;
#pragma unroll
  for (int e = 0; e < 16; ++e) {
    int r = e >> 1, bit = e & 1;
    int k = ((e & 8) ? 16 : 0) + ((r & 3) << 1) + bit + (kh << 3);
    a[e] = p[(long)(k0 + k) * kstride];
  }
  return a;
}

// A operand from f32 storage (LDS score matrix), cvt to bf16 on load.
__device__ __forceinline__ v16bf load_tileA_f32(const float* A, int ld, int m0,
                                                int k0) {
  int lane = threadIdx.x & 31;
  const float* p = A + (size_t)(m0 + (lane & 15)) * ld + k0 + ((lane >> 4) << 3);
  v16bf a;
#pragma unroll
  for (int e = 0; e < 8; ++e) a[e] = (bf16)p[e];
#pragma unroll
  for (int e = 0; e < 8; ++e) a[8 + e] = (bf16)p[16 + e];
  return a;
}

// K-loop: one A tile feeds NT B tiles (A-operand reuse).
template <int NT>
__device__ __forceinline__ void gloop(const bf16* A, int lda, const bf16* Bt,
                                      int ldb, int m0, int n0, int K, v8f* acc) {
  for (int k = 0; k < K; k += 32) {
    v16bf a = load_tileA(A, lda, m0, k);
#pragma unroll
    for (int t = 0; t < NT; ++t)
      acc[t] = wmma32(a, load_tileBt_nb(Bt, ldb, n0 + 16 * t, k), acc[t]);
  }
}

// Dual K-loop: one A tile feeds 2 B-matrices x NT tiles (4 wmma / A load).
template <int NT>
__device__ __forceinline__ void gloop2(const bf16* A, int lda, const bf16* B1t,
                                       const bf16* B2t, int ldb, int m0, int n0,
                                       int K, v8f* a1, v8f* a2) {
  for (int k = 0; k < K; k += 32) {
    v16bf a = load_tileA(A, lda, m0, k);
#pragma unroll
    for (int t = 0; t < NT; ++t) {
      a1[t] = wmma32(a, load_tileBt_nb(B1t, ldb, n0 + 16 * t, k), a1[t]);
      a2[t] = wmma32(a, load_tileBt_nb(B2t, ldb, n0 + 16 * t, k), a2[t]);
    }
  }
}

// ------------------------------------------------------------ tiny kernels ---
__global__ void evo_convT(const float* __restrict__ W, bf16* __restrict__ Wt,
                          int Kd, int Nd) {
  int idx = blockIdx.x * 256 + threadIdx.x;
  if (idx >= Kd * Nd) return;
  int n = idx / Kd, k = idx - n * Kd;
  Wt[idx] = (bf16)W[(size_t)k * Nd + n];   // W[K][N] -> Wt[N][K]
}

__global__ void evo_maskf(const unsigned char* __restrict__ sm,
                          float* __restrict__ mf) {
  int j = threadIdx.x;
  mf[j] = sm[j] ? 1.0f : 0.0f;
}

__global__ void evo_invnorm(const float* __restrict__ msa,
                            float* __restrict__ invn) {
  int i = blockIdx.x, j = threadIdx.x;
  float s = 0.0f;
  for (int t = 0; t < 64; ++t) s += msa[t * 256 + i] * msa[t * 256 + j];
  invn[i * 256 + j] = 1.0f / (s + 1e-3f);
}

// LayerNorm over last dim (blockDim.x == D, one row per block); bf16 out.
__global__ void evo_ln(const float* __restrict__ X, const float* __restrict__ G,
                       const float* __restrict__ B, bf16* __restrict__ Y) {
  __shared__ float sb[128];
  int D = blockDim.x;
  size_t row = blockIdx.x;
  int t = threadIdx.x;
  float v = X[row * D + t];
  sb[t] = v; __syncthreads();
  for (int s = D >> 1; s > 0; s >>= 1) { if (t < s) sb[t] += sb[t + s]; __syncthreads(); }
  float mu = sb[0] / D;
  __syncthreads();
  float d = v - mu;
  sb[t] = d * d; __syncthreads();
  for (int s = D >> 1; s > 0; s >>= 1) { if (t < s) sb[t] += sb[t + s]; __syncthreads(); }
  float var = sb[0] / D;
  Y[row * D + t] = (bf16)(d * rsqrtf(var + 1e-5f) * G[t] + B[t]);
}

// softmax over j of b[i,j,h] (+mask bias); out W[h][i][j] bf16 (K-contig for WMMA).
__global__ void evo_softmax_ave(const float* __restrict__ Bv,
                                const float* __restrict__ maskf,
                                bf16* __restrict__ W) {
  __shared__ float sb[256];
  int i = blockIdx.x, h = blockIdx.y, j = threadIdx.x;
  float v = Bv[((long)i * 256 + j) * 8 + h] + (maskf[j] - 1.0f) * 1e9f;
  sb[j] = v; __syncthreads();
  for (int s = 128; s > 0; s >>= 1) { if (j < s) sb[j] = fmaxf(sb[j], sb[j + s]); __syncthreads(); }
  float mx = sb[0]; __syncthreads();
  float e = __expf(v - mx);
  sb[j] = e; __syncthreads();
  for (int s = 128; s > 0; s >>= 1) { if (j < s) sb[j] += sb[j + s]; __syncthreads(); }
  W[(size_t)h * 65536 + (size_t)i * 256 + j] = (bf16)(e / sb[0]);
}

// ------------------------------------------------------------ GEMM kernels ---
// Store strides: idx = n*sn + (m>>8)*sx + (m&255)*sy  (std layout: sn=1,
// sx=256*N, sy=N -> idx = m*N + n; also used for transposed/channel layouts).
// Wave tile = 16x32 C; block = 8 waves = 64M x 64N.

__global__ __launch_bounds__(256) void evo_gemm_store(
    const bf16* __restrict__ A, int lda, const bf16* __restrict__ Bt, int ldb,
    bf16* __restrict__ C, int M, int N, int K, long sn, long sx, long sy,
    const float* __restrict__ rowscale, int act) {
  int w = threadIdx.x >> 5;
  int n0 = (blockIdx.x * 2 + (w >> 2)) * 32;
  int m0 = (blockIdx.y * 4 + (w & 3)) * 16;
  if (m0 >= M || n0 >= N) return;
  v8f acc[2] = {vz8(), vz8()};
  int nt = (n0 + 16 < N) ? 2 : 1;
  if (nt == 2) gloop<2>(A, lda, Bt, ldb, m0, n0, K, acc);
  else         gloop<1>(A, lda, Bt, ldb, m0, n0, K, acc);
  int lane = threadIdx.x & 31;
  int kh = lane >> 4;
  for (int t = 0; t < nt; ++t) {
    int cn = n0 + 16 * t + (lane & 15);
#pragma unroll
    for (int r = 0; r < 8; ++r) {
      int m = m0 + r + kh * 8;
      float v = acc[t][r];
      if (rowscale) v *= rowscale[m];
      if (act == 1) v = sigf(v);
      C[(long)cn * sn + (long)(m >> 8) * sx + (long)(m & 255) * sy] = (bf16)v;
    }
  }
}

// f32 store, bounds-checked B (used for the N=8 / N=4 bias projections).
__global__ __launch_bounds__(256) void evo_gemm_f32(
    const bf16* __restrict__ A, int lda, const bf16* __restrict__ Bt, int ldb,
    float* __restrict__ C, int M, int N, int K, long sn, long sx, long sy) {
  int w = threadIdx.x >> 5;
  int n0 = (blockIdx.x * 2 + (w >> 2)) * 16;
  int m0 = (blockIdx.y * 4 + (w & 3)) * 16;
  if (m0 >= M) return;
  v8f acc = vz8();
  for (int k = 0; k < K; k += 32)
    acc = wmma32(load_tileA(A, lda, m0, k), load_tileBt(Bt, ldb, n0, k, N), acc);
  int lane = threadIdx.x & 31;
  int cn = n0 + (lane & 15);
  if (cn >= N) return;
  int kh = lane >> 4;
#pragma unroll
  for (int r = 0; r < 8; ++r) {
    int m = m0 + r + kh * 8;
    C[(long)cn * sn + (long)(m >> 8) * sx + (long)(m & 255) * sy] = acc[r];
  }
}

// C = act(A@B1) * (A@B2) [* pairmask], act: 1=sigmoid, 2=silu. N % 64 == 0.
__global__ __launch_bounds__(256) void evo_gemm_dual(
    const bf16* __restrict__ A, int lda, const bf16* __restrict__ B1t,
    const bf16* __restrict__ B2t, int ldb, bf16* __restrict__ C, int M, int N,
    int K, long sn, long sx, long sy, const float* __restrict__ maskf, int act) {
  int w = threadIdx.x >> 5;
  int n0 = (blockIdx.x * 2 + (w >> 2)) * 32;
  int m0 = (blockIdx.y * 4 + (w & 3)) * 16;
  if (m0 >= M) return;
  v8f acc1[2] = {vz8(), vz8()}, acc2[2] = {vz8(), vz8()};
  gloop2<2>(A, lda, B1t, B2t, ldb, m0, n0, K, acc1, acc2);
  int lane = threadIdx.x & 31;
  int kh = lane >> 4;
#pragma unroll
  for (int t = 0; t < 2; ++t) {
    int cn = n0 + 16 * t + (lane & 15);
#pragma unroll
    for (int r = 0; r < 8; ++r) {
      int m = m0 + r + kh * 8;
      float u = acc1[t][r];
      float g = (act == 2) ? (u * sigf(u)) : sigf(u);
      float v = g * acc2[t][r];
      if (maskf) v *= maskf[m >> 8] * maskf[m & 255];
      C[(long)cn * sn + (long)(m >> 8) * sx + (long)(m & 255) * sy] = (bf16)v;
    }
  }
}

// C(f32) += (A@B) [* gmul]   (residual accumulation into d_out).
__global__ __launch_bounds__(256) void evo_gemm_acc(
    const bf16* __restrict__ A, int lda, const bf16* __restrict__ Bt, int ldb,
    float* __restrict__ C, const bf16* __restrict__ gmul, int M, int N, int K) {
  int w = threadIdx.x >> 5;
  int n0 = (blockIdx.x * 2 + (w >> 2)) * 32;
  int m0 = (blockIdx.y * 4 + (w & 3)) * 16;
  if (m0 >= M || n0 >= N) return;
  v8f acc[2] = {vz8(), vz8()};
  gloop<2>(A, lda, Bt, ldb, m0, n0, K, acc);
  int lane = threadIdx.x & 31;
  int kh = lane >> 4;
#pragma unroll
  for (int t = 0; t < 2; ++t) {
    int cn = n0 + 16 * t + (lane & 15);
#pragma unroll
    for (int r = 0; r < 8; ++r) {
      int m = m0 + r + kh * 8;
      long idx = (long)m * N + cn;
      float v = acc[t][r];
      if (gmul) v *= (float)gmul[idx];
      C[idx] += v;
    }
  }
}

// opm: AB[(i,jj,c,d)] = sum_s a[(i,c)][s] * b[(jj,d)][s]   M=8192 N=512 K=64
__global__ __launch_bounds__(256) void evo_gemm_opm_ab(
    const bf16* __restrict__ A, const bf16* __restrict__ Bt,
    bf16* __restrict__ AB) {
  int w = threadIdx.x >> 5;
  int n0 = (blockIdx.x * 2 + (w >> 2)) * 32;
  int m0 = (blockIdx.y * 4 + (w & 3)) * 16;
  v8f acc[2] = {vz8(), vz8()};
  gloop<2>(A, 64, Bt, 64, m0, n0, 64, acc);
  int lane = threadIdx.x & 31;
  int kh = lane >> 4;
#pragma unroll
  for (int t = 0; t < 2; ++t) {
    int n = n0 + 16 * t + (lane & 15);
#pragma unroll
    for (int r = 0; r < 8; ++r) {
      int m = m0 + r + kh * 8;  // (i,c): i=m>>5, c=m&31 ; n=(jj,d)
      AB[((((long)(m >> 5) * 16 + (n >> 5)) * 32 + (m & 31)) << 5) + (n & 31)] =
          (bf16)acc[t][r];
    }
  }
}

// opm out: Z[i, js*16+jj, :] += (AB@W + bias) * invnorm    M=4096 N=128 K=1024
__global__ __launch_bounds__(256) void evo_gemm_opm_out(
    const bf16* __restrict__ AB, const bf16* __restrict__ Wt,
    const float* __restrict__ bias, const float* __restrict__ invn,
    float* __restrict__ Z, int js) {
  int w = threadIdx.x >> 5;
  int n0 = (blockIdx.x * 2 + (w >> 2)) * 32;
  int m0 = (blockIdx.y * 4 + (w & 3)) * 16;
  v8f acc[2] = {vz8(), vz8()};
  gloop<2>(AB, 1024, Wt, 1024, m0, n0, 1024, acc);
  int lane = threadIdx.x & 31;
  int kh = lane >> 4;
#pragma unroll
  for (int t = 0; t < 2; ++t) {
    int cn = n0 + 16 * t + (lane & 15);
#pragma unroll
    for (int r = 0; r < 8; ++r) {
      int m = m0 + r + kh * 8;
      int zrow = (m >> 4) * 256 + js * 16 + (m & 15);
      Z[(long)zrow * 128 + cn] += (acc[t][r] + bias[cn]) * invn[zrow];
    }
  }
}

// triangle multiply einsum: O[i,j,c] = sum_k M1[c][i][k] * M2[c][j][k]
__global__ __launch_bounds__(128) void evo_einsum_ch(
    const bf16* __restrict__ M1, const bf16* __restrict__ M2,
    float* __restrict__ O) {
  int c = blockIdx.z;
  int w = threadIdx.x >> 5;
  int i0 = blockIdx.y * 16;
  int j0 = (blockIdx.x * 4 + w) * 32;
  const bf16* A = M1 + (size_t)c * 65536;
  const bf16* B = M2 + (size_t)c * 65536;
  v8f acc[2] = {vz8(), vz8()};
  gloop<2>(A, 256, B, 256, i0, j0, 256, acc);
  int lane = threadIdx.x & 31;
  int kh = lane >> 4;
#pragma unroll
  for (int t = 0; t < 2; ++t) {
    int j = j0 + 16 * t + (lane & 15);
#pragma unroll
    for (int r = 0; r < 8; ++r) {
      int i = i0 + r + kh * 8;
      O[((size_t)i * 256 + j) * 128 + c] = acc[t][r];
    }
  }
}

// MSA ave: O[s,i,h,c] = g * sum_j W[h][i][j] * v[s,j,h,c]
__global__ __launch_bounds__(32) void evo_ave_o(
    const bf16* __restrict__ Wsm, const bf16* __restrict__ V,
    const bf16* __restrict__ G, bf16* __restrict__ O) {
  int sh = blockIdx.z; int s = sh >> 3; int h = sh & 7;
  int i0 = blockIdx.y * 16; int c0 = blockIdx.x * 16;
  const bf16* A = Wsm + (size_t)h * 65536;             // [i][j], ld 256
  const bf16* Vb = V + (size_t)s * 65536 + h * 32;     // elem(n=c,k=j)=Vb[c+j*256]
  v8f acc = vz8();
  for (int k = 0; k < 256; k += 32)
    acc = wmma32(load_tileA(A, 256, i0, k), load_tile_gather(Vb, 1, 256, c0, k), acc);
  int lane = threadIdx.x & 31;
  int c = c0 + (lane & 15);
  int kh = lane >> 4;
#pragma unroll
  for (int r = 0; r < 8; ++r) {
    int i = i0 + r + kh * 8;
    size_t idx = ((size_t)s * 256 + i) * 256 + h * 32 + c;
    O[idx] = (bf16)((float)acc[r] * (float)G[idx]);
  }
}

// Fused triangle attention: block = (x, h); full 256x256 score tile in LDS.
// starting: x=i, rows=j ; ending: x=j, rows=i. Address = base + row*rstride + c.
__global__ __launch_bounds__(256) void evo_tri_att(
    const bf16* __restrict__ Q, const bf16* __restrict__ Kt,
    const bf16* __restrict__ V, const bf16* __restrict__ G,
    const float* __restrict__ BZ, const float* __restrict__ maskf,
    bf16* __restrict__ O, long baseMul, long rstride) {
  extern __shared__ char smem[];
  bf16* Qs = (bf16*)smem;          // 256x32
  bf16* Ks = Qs + 256 * 32;
  bf16* Vs = Ks + 256 * 32;
  float* S = (float*)(Vs + 256 * 32);  // 256x256 f32 scores
  int x = blockIdx.x, h = blockIdx.y;
  long base = (long)x * baseMul + (long)h * 32;
  int tid = threadIdx.x;
  // ---- stage Q/K/V into LDS: 1024 x 16B chunks per matrix ----
  for (int idx = tid; idx < 1024; idx += 256) {
    int r = idx >> 2, c8 = (idx & 3) << 3;
    long g = base + (long)r * rstride + c8;
#if EVO_ASYNC
    // CDNA5 async LDS-DMA: global -> LDS without VGPR round trip (ASYNCcnt).
    __builtin_amdgcn_global_load_async_to_lds_b128(
        (evo_gp)const_cast<bf16*>(Q + g),  (evo_lp)(Qs + (size_t)idx * 8), 0, 0);
    __builtin_amdgcn_global_load_async_to_lds_b128(
        (evo_gp)const_cast<bf16*>(Kt + g), (evo_lp)(Ks + (size_t)idx * 8), 0, 0);
    __builtin_amdgcn_global_load_async_to_lds_b128(
        (evo_gp)const_cast<bf16*>(V + g),  (evo_lp)(Vs + (size_t)idx * 8), 0, 0);
#else
    *(float4*)(Qs + (size_t)idx * 8) = *(const float4*)(Q + g);
    *(float4*)(Ks + (size_t)idx * 8) = *(const float4*)(Kt + g);
    *(float4*)(Vs + (size_t)idx * 8) = *(const float4*)(V + g);
#endif
  }
#if EVO_ASYNC
#if __has_builtin(__builtin_amdgcn_s_wait_asynccnt)
  __builtin_amdgcn_s_wait_asynccnt(0);
#else
  asm volatile("s_wait_asynccnt 0x0" ::: "memory");
#endif
#endif
  __syncthreads();
  int w = tid >> 5;
  int lane = tid & 31;
  int kh = lane >> 4;
  const float scale = 0.17677669529663687f;  // 1/sqrt(32)
  // ---- S = Q K^T * scale + bias (16x32 per wave iter, A reuse) ----
  for (int rt = w; rt < 16; rt += 8) {
    int r0 = rt * 16;
    v16bf qa = load_tileA(Qs, 32, r0, 0);
    for (int ct = 0; ct < 8; ++ct) {
      int c0 = ct * 32;
      v8f sa[2];
      sa[0] = wmma32(qa, load_tileBt_nb(Ks, 32, c0, 0), vz8());
      sa[1] = wmma32(qa, load_tileBt_nb(Ks, 32, c0 + 16, 0), vz8());
#pragma unroll
      for (int t = 0; t < 2; ++t) {
        int cn = c0 + 16 * t + (lane & 15);
#pragma unroll
        for (int r = 0; r < 8; ++r) {
          int row = r0 + r + kh * 8;
          float b = BZ[((long)row * 256 + cn) * 4 + h] + (maskf[cn] - 1.0f) * 1e9f;
          S[row * 256 + cn] = sa[t][r] * scale + b;
        }
      }
    }
  }
  __syncthreads();
  {  // softmax: one thread per row
    float* rowp = S + tid * 256;
    float mx = -1e30f;
    for (int k = 0; k < 256; ++k) mx = fmaxf(mx, rowp[k]);
    float sum = 0.0f;
    for (int k = 0; k < 256; ++k) { float e = __expf(rowp[k] - mx); rowp[k] = e; sum += e; }
    float inv = 1.0f / sum;
    for (int k = 0; k < 256; ++k) rowp[k] *= inv;
  }
  __syncthreads();
  // ---- O = P V (A tile reused across both 16-col output tiles) ----
  for (int rt = w; rt < 16; rt += 8) {
    int r0 = rt * 16;
    v8f acc[2] = {vz8(), vz8()};
    for (int k = 0; k < 256; k += 32) {
      v16bf a = load_tileA_f32(S, 256, r0, k);
#pragma unroll
      for (int t = 0; t < 2; ++t)
        acc[t] = wmma32(a, load_tile_gather(Vs, 1, 32, 16 * t, k), acc[t]);
    }
#pragma unroll
    for (int t = 0; t < 2; ++t) {
      int cn = 16 * t + (lane & 15);
#pragma unroll
      for (int r = 0; r < 8; ++r) {
        int row = r0 + r + kh * 8;
        long g = base + (long)row * rstride + cn;
        O[g] = (bf16)(acc[t][r] * (float)G[g]);
      }
    }
  }
}

// ================================================================== host ===
extern "C" void kernel_launch(void* const* d_in, const int* in_sizes, int n_in,
                              void* d_out, int out_size, void* d_ws,
                              size_t ws_size, hipStream_t stream) {
  (void)in_sizes; (void)n_in; (void)out_size; (void)ws_size;
  const float* m_in = (const float*)d_in[0];
  const float* z_in = (const float*)d_in[1];
  const float* msa  = (const float*)d_in[2];
  const unsigned char* smask = (const unsigned char*)d_in[3];
  auto P = [&](int i) { return (const float*)d_in[i]; };

  float* m_out = (float*)d_out;                    // [64,256,64]
  float* z_out = m_out + (size_t)64 * 256 * 64;    // [256,256,128]
  (void)hipMemcpyAsync(m_out, m_in, sizeof(float) * 1048576,
                       hipMemcpyDeviceToDevice, stream);
  (void)hipMemcpyAsync(z_out, z_in, sizeof(float) * 8388608,
                       hipMemcpyDeviceToDevice, stream);

  char* ws = (char*)d_ws; size_t off = 0;
  auto alloc = [&](size_t bytes) -> void* {
    void* p = ws + off; off += (bytes + 255) & ~(size_t)255; return p;
  };
  auto convW = [&](int idx, int Kd, int Nd) -> bf16* {  // W[K][N] -> bf16 Wt[N][K]
    bf16* dst = (bf16*)alloc((size_t)Kd * Nd * sizeof(bf16));
    int total = Kd * Nd;
    evo_convT<<<dim3((total + 255) / 256), dim3(256), 0, stream>>>(P(idx), dst, Kd, Nd);
    return dst;
  };

  // ---- weights -> transposed bf16 ----
  bf16* Wopm_a = convW(6, 64, 32);
  bf16* Wopm_b = convW(7, 64, 32);
  bf16* Wopm_o = convW(8, 1024, 128);
  bf16* Wave_v = convW(14, 64, 256);
  bf16* Wave_b = convW(15, 128, 8);
  bf16* Wave_g = convW(16, 64, 256);
  bf16* Wave_o = convW(17, 256, 64);
  bf16* Wtm_w1 = convW(20, 64, 256);
  bf16* Wtm_w2 = convW(21, 64, 256);
  bf16* Wtm_wo = convW(22, 256, 64);
  bf16* Wtmo[6]; { int ids[6] = {25, 26, 27, 28, 29, 32};
    for (int i = 0; i < 6; ++i) Wtmo[i] = convW(ids[i], 128, 128); }
  bf16* Wtmi[6]; { int ids[6] = {35, 36, 37, 38, 39, 42};
    for (int i = 0; i < 6; ++i) Wtmi[i] = convW(ids[i], 128, 128); }
  bf16* Wta[2][6];  // q,k,v,b,g,out for ta_start / ta_end
  { int b0[2] = {45, 53};
    for (int t = 0; t < 2; ++t) {
      Wta[t][0] = convW(b0[t] + 0, 128, 128);
      Wta[t][1] = convW(b0[t] + 1, 128, 128);
      Wta[t][2] = convW(b0[t] + 2, 128, 128);
      Wta[t][3] = convW(b0[t] + 3, 128, 4);
      Wta[t][4] = convW(b0[t] + 4, 128, 128);
      Wta[t][5] = convW(b0[t] + 5, 128, 128);
    } }
  bf16* Wtz1 = convW(61, 128, 512);
  bf16* Wtz2 = convW(62, 128, 512);
  bf16* Wtzo = convW(63, 512, 128);

  // ---- scratch arenas ----
  float* MASKF = (float*)alloc(256 * sizeof(float));
  float* INVN  = (float*)alloc(65536 * sizeof(float));
  float* BZF   = (float*)alloc((size_t)65536 * 8 * sizeof(float));
  bf16*  MN    = (bf16*)alloc((size_t)16384 * 64 * sizeof(bf16));
  bf16*  SA2   = (bf16*)alloc((size_t)16384 * 32 * sizeof(bf16));  // [(i,c)][s]
  bf16*  SB2   = (bf16*)alloc((size_t)16384 * 32 * sizeof(bf16));  // [(j,d)][s]
  bf16*  WSM   = (bf16*)alloc((size_t)65536 * 8 * sizeof(bf16));   // [h][i][j]
  bf16*  A0 = (bf16*)alloc((size_t)65536 * 128 * sizeof(bf16));
  bf16*  A1 = (bf16*)alloc((size_t)65536 * 128 * sizeof(bf16));
  bf16*  A2 = (bf16*)alloc((size_t)65536 * 128 * sizeof(bf16));
  bf16*  A3 = (bf16*)alloc((size_t)65536 * 128 * sizeof(bf16));
  bf16*  A4 = (bf16*)alloc((size_t)65536 * 128 * sizeof(bf16));
  float* TF32 = (float*)alloc((size_t)65536 * 128 * sizeof(float));

  evo_maskf<<<dim3(1), dim3(256), 0, stream>>>(smask, MASKF);
  evo_invnorm<<<dim3(256), dim3(256), 0, stream>>>(msa, INVN);

  auto gg64 = [](int M, int N) { return dim3((N + 63) / 64, (M + 63) / 64); };
  auto gstore = [&](const bf16* A, int lda, const bf16* Bt, int ldb, bf16* C,
                    int M, int N, int K, long sn, long sx, long sy,
                    const float* rs, int act) {
    evo_gemm_store<<<gg64(M, N), 256, 0, stream>>>(A, lda, Bt, ldb, C, M, N, K,
                                                   sn, sx, sy, rs, act);
  };
  auto gf32 = [&](const bf16* A, int lda, const bf16* Bt, int ldb, float* C,
                  int M, int N, int K, long sn, long sx, long sy) {
    evo_gemm_f32<<<dim3((N + 31) / 32, (M + 63) / 64), 256, 0, stream>>>(
        A, lda, Bt, ldb, C, M, N, K, sn, sx, sy);
  };
  auto gdual = [&](const bf16* A, int lda, const bf16* B1, const bf16* B2,
                   int ldb, bf16* C, int M, int N, int K, long sn, long sx,
                   long sy, const float* mf, int act) {
    evo_gemm_dual<<<gg64(M, N), 256, 0, stream>>>(A, lda, B1, B2, ldb, C, M, N,
                                                  K, sn, sx, sy, mf, act);
  };
  auto gacc = [&](const bf16* A, int lda, const bf16* Bt, int ldb, float* C,
                  const bf16* gm, int M, int N, int K) {
    evo_gemm_acc<<<gg64(M, N), 256, 0, stream>>>(A, lda, Bt, ldb, C, gm, M, N, K);
  };

  // ======================= 1) outer product mean -> z =======================
  evo_ln<<<dim3(16384), dim3(64), 0, stream>>>(m_out, P(4), P(5), MN);
  // a,b projections (rowscale = msa_mask), stored [(tok,c)][s] for K=s GEMM
  gstore(MN, 64, Wopm_a, 64, SA2, 16384, 32, 64, 64, 1, 2048, msa, 0);
  gstore(MN, 64, Wopm_b, 64, SB2, 16384, 32, 64, 64, 1, 2048, msa, 0);
  for (int js = 0; js < 16; ++js) {  // 16-column stripes of j
    evo_gemm_opm_ab<<<dim3(8, 128), 256, 0, stream>>>(
        SA2, SB2 + (size_t)js * 512 * 64, A1);
    evo_gemm_opm_out<<<dim3(2, 64), 256, 0, stream>>>(A1, Wopm_o, P(9), INVN,
                                                      z_out, js);
  }

  // ======================= 2) MSA pair-weighted averaging -> m ==============
  evo_ln<<<dim3(16384), dim3(64), 0, stream>>>(m_out, P(10), P(11), MN);
  gstore(MN, 64, Wave_v, 64, A4, 16384, 256, 64, 1, (long)256 * 256, 256, nullptr, 0);
  gstore(MN, 64, Wave_g, 64, A3, 16384, 256, 64, 1, (long)256 * 256, 256, nullptr, 1);
  evo_ln<<<dim3(65536), dim3(128), 0, stream>>>(z_out, P(12), P(13), A0);
  gf32(A0, 128, Wave_b, 128, BZF, 65536, 8, 128, 1, (long)8 * 256, 8);
  evo_softmax_ave<<<dim3(256, 8), 256, 0, stream>>>(BZF, MASKF, WSM);
  evo_ave_o<<<dim3(2, 16, 512), 32, 0, stream>>>(WSM, A4, A3, A1);
  gacc(A1, 256, Wave_o, 256, m_out, nullptr, 16384, 64, 256);

  // ======================= 3) MSA transition -> m ===========================
  evo_ln<<<dim3(16384), dim3(64), 0, stream>>>(m_out, P(18), P(19), MN);
  gdual(MN, 64, Wtm_w1, Wtm_w2, 64, A1, 16384, 256, 64, 1, (long)256 * 256, 256,
        nullptr, 2);
  gacc(A1, 256, Wtm_wo, 256, m_out, nullptr, 16384, 64, 256);

  // ======================= 4/5) triangle multiplications -> z ===============
  auto tri_mult = [&](const float* lnG, const float* lnB, bf16* const* W,
                      const float* loG, const float* loB, bool outgoing) {
    evo_ln<<<dim3(65536), dim3(128), 0, stream>>>(z_out, lnG, lnB, A0);
    long sx = outgoing ? 256 : 1, sy = outgoing ? 1 : 256;  // [c][x][y]/[c][y][x]
    gdual(A0, 128, W[0], W[1], 128, A1, 65536, 128, 128, 65536, sx, sy, MASKF, 1);
    gdual(A0, 128, W[2], W[3], 128, A2, 65536, 128, 128, 65536, sx, sy, MASKF, 1);
    gstore(A0, 128, W[4], 128, A3, 65536, 128, 128, 1, (long)128 * 256, 128,
           nullptr, 1);  // gate
    const bf16* M1 = outgoing ? A1 : A2;
    const bf16* M2 = outgoing ? A2 : A1;
    evo_einsum_ch<<<dim3(2, 16, 128), 128, 0, stream>>>(M1, M2, TF32);
    evo_ln<<<dim3(65536), dim3(128), 0, stream>>>(TF32, loG, loB, A0);
    gacc(A0, 128, W[5], 128, z_out, A3, 65536, 128, 128);
  };
  tri_mult(P(23), P(24), Wtmo, P(30), P(31), true);
  tri_mult(P(33), P(34), Wtmi, P(40), P(41), false);

  // ======================= 6/7) triangle attention -> z =====================
  const size_t TRI_SMEM = (size_t)3 * 256 * 32 * sizeof(bf16)
                        + (size_t)256 * 256 * sizeof(float);  // 304 KB of WGP LDS
  auto tri_attn = [&](const float* lnG, const float* lnB, bf16* const* W,
                      bool starting) {
    evo_ln<<<dim3(65536), dim3(128), 0, stream>>>(z_out, lnG, lnB, A0);
    long stdx = (long)128 * 256;
    gstore(A0, 128, W[0], 128, A1, 65536, 128, 128, 1, stdx, 128, nullptr, 0);  // q
    gstore(A0, 128, W[1], 128, A2, 65536, 128, 128, 1, stdx, 128, nullptr, 0);  // k
    gstore(A0, 128, W[2], 128, A4, 65536, 128, 128, 1, stdx, 128, nullptr, 0);  // v
    gstore(A0, 128, W[4], 128, A3, 65536, 128, 128, 1, stdx, 128, nullptr, 1);  // gate
    gf32(A0, 128, W[3], 128, BZF, 65536, 4, 128, 1, (long)4 * 256, 4);          // bz
    long baseMul = starting ? 32768 : 128;
    long rstride = starting ? 128 : 32768;
    evo_tri_att<<<dim3(256, 4), 256, TRI_SMEM, stream>>>(A1, A2, A4, A3, BZF,
                                                         MASKF, A0, baseMul,
                                                         rstride);
    gacc(A0, 128, W[5], 128, z_out, nullptr, 65536, 128, 128);
  };
  tri_attn(P(43), P(44), Wta[0], true);
  tri_attn(P(51), P(52), Wta[1], false);

  // ======================= 8) pair transition -> z ==========================
  evo_ln<<<dim3(65536), dim3(128), 0, stream>>>(z_out, P(59), P(60), A0);
  for (int ch = 0; ch < 4; ++ch) {  // 16384-row chunks (fits 16MB arena)
    const bf16* Ach = A0 + (size_t)ch * 16384 * 128;
    gdual(Ach, 128, Wtz1, Wtz2, 128, A1, 16384, 512, 128, 1, (long)512 * 256,
          512, nullptr, 2);
    gacc(A1, 512, Wtzo, 512, z_out + (size_t)ch * 16384 * 128, nullptr, 16384,
         128, 512);
  }
}
